// ClsSO3ConvModel_35682588295469
// MI455X (gfx1250) — matrix operand order repaired and checked by reading, not compiled, not run on hardware
//
#include <hip/hip_runtime.h>
#include <hip/hip_bf16.h>

#define BB 4
#define NN 4096
#define KK 32
#define C1C 128
#define CC 256
#define KS 4
#define RM 96
#define NR 128
#define PTS_ALL (KS*RM)   // 384

typedef __attribute__((ext_vector_type(16))) __bf16 v16bf;
typedef __attribute__((ext_vector_type(8)))  float  v8f;
typedef __attribute__((ext_vector_type(2)))  float  v2f;
typedef __attribute__((ext_vector_type(4)))  unsigned v4u;

// ---------- helpers ----------

__device__ inline unsigned short bf16bits(float f) {
  unsigned u = __builtin_bit_cast(unsigned, f);
  unsigned r = u + 0x7fffu + ((u >> 16) & 1u);
  return (unsigned short)(r >> 16);
}

__device__ inline float blocksum256(float v, float* red, int tid) {
  #pragma unroll
  for (int d = 16; d > 0; d >>= 1) v += __shfl_xor(v, d, 32);
  if ((tid & 31) == 0) red[tid >> 5] = v;
  __syncthreads();
  if (tid == 0) {
    float t = 0.f;
    #pragma unroll
    for (int i = 0; i < 8; ++i) t += red[i];
    red[8] = t;
  }
  __syncthreads();
  float r = red[8];
  __syncthreads();
  return r;
}

__device__ inline unsigned rotl32(unsigned x, int d) { return (x << d) | (x >> (32 - d)); }

__device__ inline void threefry2x32(unsigned k0, unsigned k1, unsigned c0, unsigned c1,
                                    unsigned* o0, unsigned* o1) {
  unsigned ks0 = k0, ks1 = k1, ks2 = k0 ^ k1 ^ 0x1BD11BDAu;
  unsigned ks[3] = {ks0, ks1, ks2};
  unsigned x0 = c0 + ks0, x1 = c1 + ks1;
  const int R0[4] = {13, 15, 26, 6};
  const int R1[4] = {17, 29, 16, 24};
  #pragma unroll
  for (int g = 0; g < 5; ++g) {
    const int* R = ((g & 1) == 0) ? R0 : R1;
    #pragma unroll
    for (int r = 0; r < 4; ++r) { x0 += x1; x1 = rotl32(x1, R[r]); x1 ^= x0; }
    x0 += ks[(g + 1) % 3];
    x1 += ks[(g + 2) % 3] + (unsigned)(g + 1);
  }
  *o0 = x0; *o1 = x1;
}

__device__ inline float erfinv_f(float x) {
  float w = -logf((1.0f - x) * (1.0f + x));
  float p;
  if (w < 5.0f) {
    w -= 2.5f;
    p = 2.81022636e-08f;
    p = fmaf(p, w, 3.43273939e-07f);
    p = fmaf(p, w, -3.5233877e-06f);
    p = fmaf(p, w, -4.39150654e-06f);
    p = fmaf(p, w, 0.00021858087f);
    p = fmaf(p, w, -0.00125372503f);
    p = fmaf(p, w, -0.00417768164f);
    p = fmaf(p, w, 0.246640727f);
    p = fmaf(p, w, 1.50140941f);
  } else {
    w = sqrtf(w) - 3.0f;
    p = -0.000200214257f;
    p = fmaf(p, w, 0.000100950558f);
    p = fmaf(p, w, 0.00134934322f);
    p = fmaf(p, w, -0.00367342844f);
    p = fmaf(p, w, 0.00573950773f);
    p = fmaf(p, w, -0.0076224613f);
    p = fmaf(p, w, 0.00943887047f);
    p = fmaf(p, w, 1.00167406f);
    p = fmaf(p, w, 2.83297682f);
  }
  return p * x;
}

// ---------- prep ----------

__global__ void k_prep(const float* __restrict__ x, float* __restrict__ xyz,
                       float* __restrict__ sqn) {
  int i = blockIdx.x * 256 + threadIdx.x;
  if (i >= BB * NN) return;
  int b = i / NN, n = i - b * NN;
  float px = x[(b * 3 + 0) * NN + n];
  float py = x[(b * 3 + 1) * NN + n];
  float pz = x[(b * 3 + 2) * NN + n];
  xyz[i * 3 + 0] = px; xyz[i * 3 + 1] = py; xyz[i * 3 + 2] = pz;
  sqn[i] = px * px + py * py + pz * pz;
}

// transpose + bf16-quantize weights: W2 (131x256 -> [256][160] padded), Wk/Wv (256x256 -> [d][c])
__global__ void k_prepw(const float* __restrict__ W2, const float* __restrict__ Wk,
                        const float* __restrict__ Wv, unsigned short* __restrict__ W2T,
                        unsigned short* __restrict__ WkT, unsigned short* __restrict__ WvT) {
  int idx = blockIdx.x * 256 + threadIdx.x;
  if (idx < 256 * 160) {
    int d = idx / 160, c = idx - d * 160;
    float v = (c < 131) ? W2[c * CC + d] : 0.0f;
    W2T[idx] = bf16bits(v);
  } else if (idx < 256 * 160 + 65536) {
    int i2 = idx - 256 * 160; int d = i2 >> 8, c = i2 & 255;
    WkT[i2] = bf16bits(Wk[c * CC + d]);
  } else if (idx < 256 * 160 + 131072) {
    int i2 = idx - 256 * 160 - 65536; int d = i2 >> 8, c = i2 & 255;
    WvT[i2] = bf16bits(Wv[c * CC + d]);
  }
}

// ---------- KNN (brute force, register top-32) ----------

__global__ void __launch_bounds__(128) k_knn(const float* __restrict__ xyz,
                                             const float* __restrict__ sqn,
                                             int* __restrict__ knn) {
  __shared__ float tx[128], ty[128], tz[128], ts[128];
  int b = blockIdx.y;
  int q = blockIdx.x * 128 + threadIdx.x;
  int gq = b * NN + q;
  float qx = xyz[gq * 3 + 0], qy = xyz[gq * 3 + 1], qz = xyz[gq * 3 + 2];
  float qs = sqn[gq];
  float bd[KK]; int bi[KK];
  #pragma unroll
  for (int i = 0; i < KK; ++i) { bd[i] = 1e30f; bi[i] = 0; }
  for (int t = 0; t < NN / 128; ++t) {
    __syncthreads();
    int gc = b * NN + t * 128 + threadIdx.x;
    tx[threadIdx.x] = xyz[gc * 3 + 0];
    ty[threadIdx.x] = xyz[gc * 3 + 1];
    tz[threadIdx.x] = xyz[gc * 3 + 2];
    ts[threadIdx.x] = sqn[gc];
    if (t + 1 < NN / 128) {
      __builtin_prefetch(&xyz[(gc + 128) * 3], 0, 1);  // global_prefetch_b8 next tile
    }
    __syncthreads();
    for (int j = 0; j < 128; ++j) {
      float d = qs + ts[j] - 2.0f * (qx * tx[j] + qy * ty[j] + qz * tz[j]);
      if (d < bd[KK - 1]) {
        bd[KK - 1] = d; bi[KK - 1] = t * 128 + j;
        #pragma unroll
        for (int s = KK - 1; s >= 1; --s) {
          if (bd[s] < bd[s - 1]) {
            float td = bd[s]; bd[s] = bd[s - 1]; bd[s - 1] = td;
            int ti = bi[s]; bi[s] = bi[s - 1]; bi[s - 1] = ti;
          }
        }
      }
    }
  }
  #pragma unroll
  for (int i = 0; i < KK; ++i) knn[(size_t)gq * KK + i] = bi[i];
}

// ---------- conv block 1: WMMA f32 16x16x4 (K = 4 features exactly) ----------

__global__ void __launch_bounds__(256) k_conv1(const float* __restrict__ xyz,
                                               const int* __restrict__ knn,
                                               const float* __restrict__ W1,
                                               const float* __restrict__ b1,
                                               float* __restrict__ f1) {
  int wid = blockIdx.x * 8 + (threadIdx.x >> 5);
  int lane = threadIdx.x & 31;
  int hf = lane >> 4, mr = lane & 15;
  int pt = wid;
  int b = pt / NN;
  float cx = xyz[pt * 3 + 0], cy = xyz[pt * 3 + 1], cz = xyz[pt * 3 + 2];
  v2f a0, a1;
  {
    int nb = knn[(size_t)pt * KK + mr];
    int g = b * NN + nb;
    float rx = xyz[g * 3 + 0] - cx, ry = xyz[g * 3 + 1] - cy, rz = xyz[g * 3 + 2] - cz;
    a0.x = hf ? ry : 1.0f;     // element e=0 -> feature 2*hf
    a0.y = hf ? rz : rx;       // element e=1 -> feature 2*hf+1
  }
  {
    int nb = knn[(size_t)pt * KK + 16 + mr];
    int g = b * NN + nb;
    float rx = xyz[g * 3 + 0] - cx, ry = xyz[g * 3 + 1] - cy, rz = xyz[g * 3 + 2] - cz;
    a1.x = hf ? ry : 1.0f;
    a1.y = hf ? rz : rx;
  }
  v8f zero = {0.f, 0.f, 0.f, 0.f, 0.f, 0.f, 0.f, 0.f};
  for (int t = 0; t < 8; ++t) {
    int col = 16 * t + mr;
    v2f bf;
    bf.x = W1[(2 * hf + 0) * C1C + col];
    bf.y = W1[(2 * hf + 1) * C1C + col];
    v8f d0 = __builtin_amdgcn_wmma_f32_16x16x4_f32(false, a0, false, bf, (short)0, zero, false, false);
    v8f d1 = __builtin_amdgcn_wmma_f32_16x16x4_f32(false, a1, false, bf, (short)0, zero, false, false);
    float bias = b1[col];
    float m = 0.0f;
    #pragma unroll
    for (int r = 0; r < 8; ++r) {
      m = fmaxf(m, fmaxf(d0[r] + bias, 0.0f));
      m = fmaxf(m, fmaxf(d1[r] + bias, 0.0f));
    }
    m = fmaxf(m, __shfl_xor(m, 16, 32));
    if (lane < 16) f1[(size_t)pt * C1C + col] = m;
  }
}

// ---------- conv block 2: WMMA bf16 16x16x32, [32 x 160pad] x [160 x 256] per point ----------

__global__ void __launch_bounds__(256) k_conv2(const float* __restrict__ xyz,
                                               const int* __restrict__ knn,
                                               const float* __restrict__ f1,
                                               const unsigned short* __restrict__ W2T,
                                               const float* __restrict__ b2,
                                               float* __restrict__ f2) {
  __shared__ __align__(16) unsigned short sa[2][32][160];
  __shared__ int sk[2][32];
  __shared__ float sc[2][3];
  int tid = threadIdx.x;
  int ptb = blockIdx.x * 2;
  int b = ptb / NN;
  if (tid < 64) {
    int pi = tid >> 5, row = tid & 31;
    sk[pi][row] = knn[(size_t)(ptb + pi) * KK + row];
  }
  if (tid < 6) {
    int pi = tid / 3, d = tid - pi * 3;
    sc[pi][d] = xyz[(ptb + pi) * 3 + d];
  }
  __syncthreads();
  for (int idx = tid; idx < 2 * 32 * 160; idx += 256) {
    int pi = idx / (32 * 160);
    int rem = idx - pi * 32 * 160;
    int row = rem / 160, c = rem - row * 160;
    int nb = sk[pi][row];
    int g = b * NN + nb;
    float v;
    if (c < C1C)          v = f1[(size_t)g * C1C + c];
    else if (c < C1C + 3) v = xyz[g * 3 + (c - C1C)] - sc[pi][c - C1C];
    else                  v = 0.0f;
    sa[pi][row][c] = bf16bits(v);
  }
  __syncthreads();
  int w = tid >> 5, lane = tid & 31;
  int hf = lane >> 4, mr = lane & 15;
  int pi = w >> 2;
  int colg = (w & 3) * 64;
  int pt = ptb + pi;
  v8f zero = {0.f, 0.f, 0.f, 0.f, 0.f, 0.f, 0.f, 0.f};
  v8f acc[2][4];
  #pragma unroll
  for (int mt = 0; mt < 2; ++mt)
    #pragma unroll
    for (int nt = 0; nt < 4; ++nt) acc[mt][nt] = zero;
  for (int ksi = 0; ksi < 5; ++ksi) {
    int kb = ksi * 32;
    union { v16bf v; v4u q[2]; } A[2];
    #pragma unroll
    for (int mt = 0; mt < 2; ++mt) {
      // A layout (16-bit, 16x32): u[0..3] <- k = kb+8*hf .. +7 ; u[4..7] <- k = kb+16+8*hf .. +7
      const unsigned short* rp = &sa[pi][16 * mt + mr][0];
      A[mt].q[0] = *(const v4u*)(rp + kb + 8 * hf);
      A[mt].q[1] = *(const v4u*)(rp + kb + 16 + 8 * hf);
    }
    #pragma unroll
    for (int nt = 0; nt < 4; ++nt) {
      int col = colg + 16 * nt + mr;
      union { v16bf v; v4u q[2]; } Bf;
      // B layout (16-bit, 32x16): u[0..3] <- k = kb+8*hf .. +7 ; u[4..7] <- k = kb+16+8*hf .. +7
      const unsigned short* wp = W2T + (size_t)col * 160 + kb + 8 * hf;
      Bf.q[0] = *(const v4u*)(wp);
      Bf.q[1] = *(const v4u*)(wp + 16);
      acc[0][nt] = __builtin_amdgcn_wmma_f32_16x16x32_bf16(false, A[0].v, false, Bf.v, (short)0, acc[0][nt], false, false);
      acc[1][nt] = __builtin_amdgcn_wmma_f32_16x16x32_bf16(false, A[1].v, false, Bf.v, (short)0, acc[1][nt], false, false);
    }
  }
  #pragma unroll
  for (int nt = 0; nt < 4; ++nt) {
    int col = colg + 16 * nt + mr;
    float bias = b2[col];
    float m = 0.0f;
    #pragma unroll
    for (int r = 0; r < 8; ++r) {
      m = fmaxf(m, fmaxf(acc[0][nt][r] + bias, 0.0f));
      m = fmaxf(m, fmaxf(acc[1][nt][r] + bias, 0.0f));
    }
    m = fmaxf(m, __shfl_xor(m, 16, 32));
    if (lane < 16) f2[(size_t)pt * CC + col] = m;
  }
}

// ---------- LayerNorm over channels -> bf16 ----------

__global__ void __launch_bounds__(256) k_ln(const float* __restrict__ f2,
                                            const float* __restrict__ g,
                                            const float* __restrict__ bb,
                                            unsigned short* __restrict__ inpbf) {
  __shared__ float red[9];
  int pt = blockIdx.x, tid = threadIdx.x;
  float x = f2[(size_t)pt * CC + tid];
  float m = blocksum256(x, red, tid) * (1.0f / CC);
  float dv = x - m;
  float var = blocksum256(dv * dv, red, tid) * (1.0f / CC);
  float y = dv * rsqrtf(var + 1e-5f) * g[tid] + bb[tid];
  inpbf[(size_t)pt * CC + tid] = bf16bits(y);
}

// ---------- kk/vv projections: WMMA bf16, 32 points x 256 per block ----------

__global__ void __launch_bounds__(256) k_kv(const unsigned short* __restrict__ inpbf,
                                            const unsigned short* __restrict__ WkT,
                                            const unsigned short* __restrict__ WvT,
                                            float* __restrict__ kk, float* __restrict__ vvo) {
  __shared__ __align__(16) unsigned short sA[32][256];
  int tid = threadIdx.x;
  int pb = blockIdx.x * 32;
  {
    // 32x256 ushort = 4096 dwords; copy as 32-bit words
    const unsigned* src = (const unsigned*)(inpbf + (size_t)pb * CC);
    unsigned* dst = (unsigned*)&sA[0][0];
    for (int idx = tid; idx < 32 * 256 / 2; idx += 256) dst[idx] = src[idx];
  }
  __syncthreads();
  int w = tid >> 5, lane = tid & 31, hf = lane >> 4, mr = lane & 15;
  int colb = w * 32;
  v8f zero = {0.f, 0.f, 0.f, 0.f, 0.f, 0.f, 0.f, 0.f};
  v8f ak[2][2], av[2][2];
  #pragma unroll
  for (int mt = 0; mt < 2; ++mt)
    #pragma unroll
    for (int nt = 0; nt < 2; ++nt) { ak[mt][nt] = zero; av[mt][nt] = zero; }
  for (int ksi = 0; ksi < 8; ++ksi) {
    int kb = ksi * 32;
    union { v16bf v; v4u q[2]; } A[2];
    #pragma unroll
    for (int mt = 0; mt < 2; ++mt) {
      const unsigned short* rp = &sA[16 * mt + mr][0];
      A[mt].q[0] = *(const v4u*)(rp + kb + 8 * hf);
      A[mt].q[1] = *(const v4u*)(rp + kb + 16 + 8 * hf);
    }
    #pragma unroll
    for (int nt = 0; nt < 2; ++nt) {
      int col = colb + 16 * nt + mr;
      union { v16bf v; v4u q[2]; } Bk, Bv;
      const unsigned short* wpk = WkT + (size_t)col * 256 + kb + 8 * hf;
      const unsigned short* wpv = WvT + (size_t)col * 256 + kb + 8 * hf;
      Bk.q[0] = *(const v4u*)(wpk);
      Bk.q[1] = *(const v4u*)(wpk + 16);
      Bv.q[0] = *(const v4u*)(wpv);
      Bv.q[1] = *(const v4u*)(wpv + 16);
      ak[0][nt] = __builtin_amdgcn_wmma_f32_16x16x32_bf16(false, A[0].v, false, Bk.v, (short)0, ak[0][nt], false, false);
      ak[1][nt] = __builtin_amdgcn_wmma_f32_16x16x32_bf16(false, A[1].v, false, Bk.v, (short)0, ak[1][nt], false, false);
      av[0][nt] = __builtin_amdgcn_wmma_f32_16x16x32_bf16(false, A[0].v, false, Bv.v, (short)0, av[0][nt], false, false);
      av[1][nt] = __builtin_amdgcn_wmma_f32_16x16x32_bf16(false, A[1].v, false, Bv.v, (short)0, av[1][nt], false, false);
    }
  }
  #pragma unroll
  for (int mt = 0; mt < 2; ++mt)
    #pragma unroll
    for (int nt = 0; nt < 2; ++nt) {
      int col = colb + 16 * nt + mr;
      #pragma unroll
      for (int r = 0; r < 8; ++r) {
        int prow = pb + 16 * mt + 8 * hf + r;
        kk[(size_t)prow * CC + col] = ak[mt][nt][r];
        vvo[(size_t)prow * CC + col] = av[mt][nt][r];
      }
    }
}

// ---------- slots init: JAX threefry2x32 + Giles erfinv ----------

__global__ void k_slots_init(const float* __restrict__ mu, const float* __restrict__ sigma,
                             float* __restrict__ slots) {
  int i = blockIdx.x * 256 + threadIdx.x;  // < 4096
  unsigned c0, c1; bool hi;
  if (i < 2048) { c0 = (unsigned)i; c1 = (unsigned)(i + 2048); hi = false; }
  else          { c0 = (unsigned)(i - 2048); c1 = (unsigned)i; hi = true; }
  unsigned o0, o1;
  threefry2x32(0u, 42u, c0, c1, &o0, &o1);
  unsigned bits = hi ? o1 : o0;
  unsigned fb = (bits >> 9) | 0x3f800000u;
  float u01 = __builtin_bit_cast(float, fb) - 1.0f;
  const float lo = -0.99999994f;
  float u = u01 * (1.0f - lo) + lo;
  u = fmaxf(u, lo);
  float nrm = 1.41421354f * erfinv_f(u);
  int c = i & 255;
  slots[i] = mu[c] + sigma[c] * nrm;
}

// ---------- slot attention ----------

__global__ void __launch_bounds__(256) k_q(const float* __restrict__ slots,
                                           const float* __restrict__ gs, const float* __restrict__ bs,
                                           const float* __restrict__ Wq,
                                           float* __restrict__ q, float* __restrict__ asum) {
  __shared__ float red[9];
  __shared__ float hln[CC];
  int bk = blockIdx.x, tid = threadIdx.x;
  float x = slots[bk * CC + tid];
  float m = blocksum256(x, red, tid) * (1.0f / CC);
  float dv = x - m;
  float var = blocksum256(dv * dv, red, tid) * (1.0f / CC);
  hln[tid] = dv * rsqrtf(var + 1e-5f) * gs[tid] + bs[tid];
  __syncthreads();
  float acc = 0.f;
  for (int c = 0; c < CC; ++c) acc += hln[c] * Wq[c * CC + tid];
  q[bk * CC + tid] = acc;
  if (tid == 0) asum[bk] = 0.f;
}

__global__ void __launch_bounds__(256) k_attn(const float* __restrict__ kk,
                                              const float* __restrict__ q,
                                              float* __restrict__ attn,
                                              float* __restrict__ asum) {
  __shared__ float sq[KS * CC];
  __shared__ float red[9];
  int tid = threadIdx.x;
  int b = blockIdx.y;
  int n = blockIdx.x * 256 + tid;
  for (int idx = tid; idx < KS * CC; idx += 256) sq[idx] = q[b * KS * CC + idx];
  __syncthreads();
  const float* kr = kk + (size_t)(b * NN + n) * CC;
  float lg[KS] = {0.f, 0.f, 0.f, 0.f};
  for (int c = 0; c < CC; ++c) {
    float kv = kr[c];
    #pragma unroll
    for (int k = 0; k < KS; ++k) lg[k] += kv * sq[k * CC + c];
  }
  float mx = -1e30f;
  #pragma unroll
  for (int k = 0; k < KS; ++k) { lg[k] *= 0.0625f; mx = fmaxf(mx, lg[k]); }
  float e[KS], ssum = 0.f;
  #pragma unroll
  for (int k = 0; k < KS; ++k) { e[k] = expf(lg[k] - mx); ssum += e[k]; }
  float av[KS];
  #pragma unroll
  for (int k = 0; k < KS; ++k) {
    av[k] = e[k] / ssum;
    attn[(size_t)(b * KS + k) * NN + n] = av[k];
  }
  for (int k = 0; k < KS; ++k) {
    float s = blocksum256(av[k], red, tid);
    if (tid == 0) atomicAdd(&asum[b * KS + k], s);
  }
}

__global__ void __launch_bounds__(256) k_upd(const float* __restrict__ attn,
                                             const float* __restrict__ vv,
                                             const float* __restrict__ asum,
                                             float* __restrict__ upd) {
  __shared__ float sA[256];
  int bk = blockIdx.x, tid = threadIdx.x;
  int b = bk >> 2;
  float acc = 0.f;
  for (int nb = 0; nb < NN; nb += 256) {
    __syncthreads();
    sA[tid] = attn[(size_t)bk * NN + nb + tid];
    __syncthreads();
    for (int j = 0; j < 256; ++j)
      acc += sA[j] * vv[(size_t)(b * NN + nb + j) * CC + tid];
  }
  upd[bk * CC + tid] = acc / (asum[bk] + 1e-8f);
}

__global__ void __launch_bounds__(256) k_gru(const float* __restrict__ upd, float* __restrict__ slots,
                                             const float* __restrict__ W_ih, const float* __restrict__ W_hh,
                                             const float* __restrict__ b_ih, const float* __restrict__ b_hh,
                                             const float* __restrict__ gff, const float* __restrict__ bff,
                                             const float* __restrict__ Wm1, const float* __restrict__ bm1,
                                             const float* __restrict__ Wm2, const float* __restrict__ bm2) {
  __shared__ float su[CC], sp[CC], hln[CC], hb[CC];
  __shared__ float red[9];
  int bk = blockIdx.x, t = threadIdx.x;
  su[t] = upd[bk * CC + t];
  sp[t] = slots[bk * CC + t];
  __syncthreads();
  float gir = b_ih[t], giz = b_ih[CC + t], gin = b_ih[2 * CC + t];
  float ghr = b_hh[t], ghz = b_hh[CC + t], ghn = b_hh[2 * CC + t];
  for (int c = 0; c < CC; ++c) {
    float uc = su[c], pc = sp[c];
    const float* wi = W_ih + (size_t)c * 3 * CC;
    const float* wh = W_hh + (size_t)c * 3 * CC;
    gir += uc * wi[t]; giz += uc * wi[CC + t]; gin += uc * wi[2 * CC + t];
    ghr += pc * wh[t]; ghz += pc * wh[CC + t]; ghn += pc * wh[2 * CC + t];
  }
  float r = 1.f / (1.f + expf(-(gir + ghr)));
  float z = 1.f / (1.f + expf(-(giz + ghz)));
  float nn_ = tanhf(gin + r * ghn);
  float sval = (1.f - z) * nn_ + z * sp[t];
  __syncthreads();
  float m = blocksum256(sval, red, t) * (1.0f / CC);
  float dv = sval - m;
  float var = blocksum256(dv * dv, red, t) * (1.0f / CC);
  hln[t] = dv * rsqrtf(var + 1e-5f) * gff[t] + bff[t];
  __syncthreads();
  float a1 = bm1[t];
  for (int c = 0; c < CC; ++c) a1 += hln[c] * Wm1[c * CC + t];
  hb[t] = fmaxf(a1, 0.f);
  __syncthreads();
  float a2 = bm2[t];
  for (int c = 0; c < CC; ++c) a2 += hb[c] * Wm2[c * CC + t];
  slots[bk * CC + t] = sval + a2;
}

// ---------- reconstruction ----------

__global__ void __launch_bounds__(256) k_mm256(const float* __restrict__ in, const float* __restrict__ W,
                                               const float* __restrict__ bias, float* __restrict__ out) {
  __shared__ float s[CC];
  int bk = blockIdx.x, t = threadIdx.x;
  s[t] = in[bk * CC + t];
  __syncthreads();
  float a = bias[t];
  for (int c = 0; c < CC; ++c) a += s[c] * W[c * CC + t];
  out[bk * CC + t] = a;
}

__global__ void k_bnrelu(const float* __restrict__ in, const float* __restrict__ g,
                         const float* __restrict__ be, float* __restrict__ out) {
  int t = threadIdx.x;
  float m = 0.f;
  for (int i = 0; i < BB * KS; ++i) m += in[i * CC + t];
  m *= (1.f / (BB * KS));
  float v = 0.f;
  for (int i = 0; i < BB * KS; ++i) { float d = in[i * CC + t] - m; v += d * d; }
  v *= (1.f / (BB * KS));
  float sc = rsqrtf(v + 1e-5f) * g[t];
  for (int i = 0; i < BB * KS; ++i)
    out[i * CC + t] = fmaxf((in[i * CC + t] - m) * sc + be[t], 0.f);
}

__global__ void k_mm3(const float* __restrict__ in, const float* __restrict__ W,
                      const float* __restrict__ bias, float* __restrict__ out) {
  __shared__ float s[CC];
  int bk = blockIdx.x, t = threadIdx.x;  // blockDim = 288
  if (t < CC) s[t] = in[bk * CC + t];
  __syncthreads();
  if (t < RM * 3) {
    float a = bias[t];
    for (int c = 0; c < CC; ++c) a += s[c] * W[c * (RM * 3) + t];
    out[bk * (RM * 3) + t] = a;
  }
}

// ---------- FPS ----------

__global__ void __launch_bounds__(512) k_fps(const float* __restrict__ apts, float* __restrict__ ds) {
  __shared__ float px[PTS_ALL], py[PTS_ALL], pz[PTS_ALL], dd[PTS_ALL];
  __shared__ float rv[512];
  __shared__ int ri[512];
  __shared__ int slast;
  int b = blockIdx.x, t = threadIdx.x;
  if (t < PTS_ALL) {
    px[t] = apts[(b * PTS_ALL + t) * 3 + 0];
    py[t] = apts[(b * PTS_ALL + t) * 3 + 1];
    pz[t] = apts[(b * PTS_ALL + t) * 3 + 2];
    dd[t] = 1e10f;
  }
  if (t == 0) {
    slast = 0;
    ds[(b * NR + 0) * 3 + 0] = apts[b * PTS_ALL * 3 + 0];
    ds[(b * NR + 0) * 3 + 1] = apts[b * PTS_ALL * 3 + 1];
    ds[(b * NR + 0) * 3 + 2] = apts[b * PTS_ALL * 3 + 2];
  }
  __syncthreads();
  for (int i = 1; i < NR; ++i) {
    int last = slast;
    if (t < PTS_ALL) {
      float dx = px[t] - px[last], dy = py[t] - py[last], dz = pz[t] - pz[last];
      float d2 = fminf(dd[t], dx * dx + dy * dy + dz * dz);
      dd[t] = d2;
      rv[t] = d2; ri[t] = t;
    } else { rv[t] = -1e30f; ri[t] = t; }
    __syncthreads();
    for (int s = 256; s >= 1; s >>= 1) {
      if (t < s) {
        if (rv[t + s] > rv[t] || (rv[t + s] == rv[t] && ri[t + s] < ri[t])) {
          rv[t] = rv[t + s]; ri[t] = ri[t + s];
        }
      }
      __syncthreads();
    }
    int nxt = ri[0];
    if (t == 0) {
      ds[(b * NR + i) * 3 + 0] = px[nxt];
      ds[(b * NR + i) * 3 + 1] = py[nxt];
      ds[(b * NR + i) * 3 + 2] = pz[nxt];
      slast = nxt;
    }
    __syncthreads();
  }
}

// ---------- chamfer ----------

__global__ void __launch_bounds__(128) k_cham1(const float* __restrict__ ds, const float* __restrict__ xyz,
                                               float* __restrict__ dsum1) {
  __shared__ float sx[128], sy[128], sz[128];
  __shared__ float red[4];
  int b = blockIdx.x, t = threadIdx.x;
  float ax = ds[(b * NR + t) * 3 + 0];
  float ay = ds[(b * NR + t) * 3 + 1];
  float az = ds[(b * NR + t) * 3 + 2];
  float mn = 1e30f;
  for (int cb = 0; cb < NN; cb += 128) {
    __syncthreads();
    int g = b * NN + cb + t;
    sx[t] = xyz[g * 3 + 0]; sy[t] = xyz[g * 3 + 1]; sz[t] = xyz[g * 3 + 2];
    __syncthreads();
    for (int j = 0; j < 128; ++j) {
      float dx = ax - sx[j], dy = ay - sy[j], dz = az - sz[j];
      mn = fminf(mn, dx * dx + dy * dy + dz * dz);
    }
  }
  float s = mn;
  #pragma unroll
  for (int d = 16; d > 0; d >>= 1) s += __shfl_xor(s, d, 32);
  if ((t & 31) == 0) red[t >> 5] = s;
  __syncthreads();
  if (t == 0) dsum1[b] = red[0] + red[1] + red[2] + red[3];
}

__global__ void __launch_bounds__(256) k_cham2(const float* __restrict__ ds, const float* __restrict__ xyz,
                                               float* __restrict__ dsum2) {
  __shared__ float sx[NR], sy[NR], sz[NR];
  __shared__ float red[9];
  int b = blockIdx.y, t = threadIdx.x;
  int n = blockIdx.x * 256 + t;
  if (t < NR) {
    sx[t] = ds[(b * NR + t) * 3 + 0];
    sy[t] = ds[(b * NR + t) * 3 + 1];
    sz[t] = ds[(b * NR + t) * 3 + 2];
  }
  __syncthreads();
  float ax = xyz[(b * NN + n) * 3 + 0];
  float ay = xyz[(b * NN + n) * 3 + 1];
  float az = xyz[(b * NN + n) * 3 + 2];
  float mn = 1e30f;
  for (int j = 0; j < NR; ++j) {
    float dx = ax - sx[j], dy = ay - sy[j], dz = az - sz[j];
    mn = fminf(mn, dx * dx + dy * dy + dz * dz);
  }
  float s = blocksum256(mn, red, t);
  if (t == 0) atomicAdd(&dsum2[b], s);
}

__global__ void k_final(const float* __restrict__ dsum1, const float* __restrict__ dsum2,
                        float* __restrict__ out) {
  if (threadIdx.x == 0 && blockIdx.x == 0) {
    float l = 0.f;
    for (int b = 0; b < BB; ++b) l += dsum1[b] * (1.0f / NR) + dsum2[b] * (1.0f / NN);
    out[0] = l * (1.0f / BB);
  }
}

__global__ void k_attn_out(const float* __restrict__ attn, float* __restrict__ out) {
  int idx = blockIdx.x * 256 + threadIdx.x;  // B*N*KS = 65536
  int k = idx & 3;
  int n = (idx >> 2) & (NN - 1);
  int b = idx >> 14;
  out[1 + idx] = attn[(size_t)(b * KS + k) * NN + n];
}

// ---------- launch ----------

extern "C" void kernel_launch(void* const* d_in, const int* in_sizes, int n_in,
                              void* d_out, int out_size, void* d_ws, size_t ws_size,
                              hipStream_t stream) {
  (void)in_sizes; (void)n_in; (void)out_size; (void)ws_size;
  const float* x      = (const float*)d_in[0];
  const float* W1     = (const float*)d_in[2];
  const float* b1     = (const float*)d_in[3];
  const float* W2     = (const float*)d_in[4];
  const float* b2     = (const float*)d_in[5];
  const float* mu     = (const float*)d_in[6];
  const float* sigma  = (const float*)d_in[7];
  const float* lnin_g = (const float*)d_in[8];
  const float* lnin_b = (const float*)d_in[9];
  const float* lns_g  = (const float*)d_in[10];
  const float* lns_b  = (const float*)d_in[11];
  const float* lnf_g  = (const float*)d_in[12];
  const float* lnf_b  = (const float*)d_in[13];
  const float* Wq     = (const float*)d_in[14];
  const float* Wk     = (const float*)d_in[15];
  const float* Wv     = (const float*)d_in[16];
  const float* W_ih   = (const float*)d_in[17];
  const float* W_hh   = (const float*)d_in[18];
  const float* b_ih   = (const float*)d_in[19];
  const float* b_hh   = (const float*)d_in[20];
  const float* Wm1    = (const float*)d_in[21];
  const float* bm1    = (const float*)d_in[22];
  const float* Wm2    = (const float*)d_in[23];
  const float* bm2    = (const float*)d_in[24];
  const float* Wr1    = (const float*)d_in[25];
  const float* br1    = (const float*)d_in[26];
  const float* g1     = (const float*)d_in[27];
  const float* be1    = (const float*)d_in[28];
  const float* Wr2    = (const float*)d_in[29];
  const float* br2    = (const float*)d_in[30];
  const float* g2     = (const float*)d_in[31];
  const float* be2    = (const float*)d_in[32];
  const float* Wr3    = (const float*)d_in[33];
  const float* br3    = (const float*)d_in[34];

  char* wsb = (char*)d_ws;
  size_t off = 0;
  auto take = [&](size_t bytes) -> void* {
    void* p = wsb + off;
    off += (bytes + 255) & ~(size_t)255;
    return p;
  };
  float* xyz   = (float*)take((size_t)BB * NN * 3 * 4);
  float* sqn   = (float*)take((size_t)BB * NN * 4);
  int*   knn   = (int*)take((size_t)BB * NN * KK * 4);
  float* f1    = (float*)take((size_t)BB * NN * C1C * 4);
  float* f2    = (float*)take((size_t)BB * NN * CC * 4);
  unsigned short* inpbf = (unsigned short*)take((size_t)BB * NN * CC * 2);
  float* kk    = (float*)take((size_t)BB * NN * CC * 4);
  float* vv    = (float*)take((size_t)BB * NN * CC * 4);
  unsigned short* W2T = (unsigned short*)take(256 * 160 * 2);
  unsigned short* WkT = (unsigned short*)take(256 * 256 * 2);
  unsigned short* WvT = (unsigned short*)take(256 * 256 * 2);
  float* slots = (float*)take(16 * CC * 4);
  float* qbuf  = (float*)take(16 * CC * 4);
  float* attn  = (float*)take((size_t)BB * KS * NN * 4);
  float* asum  = (float*)take(16 * 4);
  float* updb  = (float*)take(16 * CC * 4);
  float* t1    = (float*)take(16 * CC * 4);
  float* h1    = (float*)take(16 * CC * 4);
  float* h2    = (float*)take(16 * CC * 4);
  float* apts  = (float*)take((size_t)BB * PTS_ALL * 3 * 4);
  float* dsp   = (float*)take((size_t)BB * NR * 3 * 4);
  float* dsum1 = (float*)take(BB * 4);
  float* dsum2 = (float*)take(BB * 4);

  float* out = (float*)d_out;

  hipMemsetAsync(dsum1, 0, BB * 4, stream);
  hipMemsetAsync(dsum2, 0, BB * 4, stream);

  k_prep<<<(BB * NN + 255) / 256, 256, 0, stream>>>(x, xyz, sqn);
  k_prepw<<<(256 * 160 + 2 * 65536 + 255) / 256, 256, 0, stream>>>(W2, Wk, Wv, W2T, WkT, WvT);
  k_knn<<<dim3(NN / 128, BB), 128, 0, stream>>>(xyz, sqn, knn);
  k_conv1<<<BB * NN / 8, 256, 0, stream>>>(xyz, knn, W1, b1, f1);
  k_conv2<<<BB * NN / 2, 256, 0, stream>>>(xyz, knn, f1, W2T, b2, f2);
  k_ln<<<BB * NN, 256, 0, stream>>>(f2, lnin_g, lnin_b, inpbf);
  k_kv<<<BB * NN / 32, 256, 0, stream>>>(inpbf, WkT, WvT, kk, vv);
  k_slots_init<<<16, 256, 0, stream>>>(mu, sigma, slots);
  for (int it = 0; it < 3; ++it) {
    k_q<<<16, 256, 0, stream>>>(slots, lns_g, lns_b, Wq, qbuf, asum);
    k_attn<<<dim3(NN / 256, BB), 256, 0, stream>>>(kk, qbuf, attn, asum);
    k_upd<<<16, 256, 0, stream>>>(attn, vv, asum, updb);
    k_gru<<<16, 256, 0, stream>>>(updb, slots, W_ih, W_hh, b_ih, b_hh,
                                  lnf_g, lnf_b, Wm1, bm1, Wm2, bm2);
  }
  k_mm256<<<16, 256, 0, stream>>>(slots, Wr1, br1, t1);
  k_bnrelu<<<1, 256, 0, stream>>>(t1, g1, be1, h1);
  k_mm256<<<16, 256, 0, stream>>>(h1, Wr2, br2, t1);
  k_bnrelu<<<1, 256, 0, stream>>>(t1, g2, be2, h2);
  k_mm3<<<16, 288, 0, stream>>>(h2, Wr3, br3, apts);
  k_fps<<<BB, 512, 0, stream>>>(apts, dsp);
  k_cham1<<<BB, 128, 0, stream>>>(dsp, xyz, dsum1);
  k_cham2<<<dim3(NN / 256, BB), 256, 0, stream>>>(dsp, xyz, dsum2);
  k_final<<<1, 1, 0, stream>>>(dsum1, dsum2, out);
  k_attn_out<<<BB * NN * KS / 256, 256, 0, stream>>>(attn, out);
}
